// SPGReservoirActor_70076686401853
// MI455X (gfx1250) — matrix operand order repaired and checked by reading, not compiled, not run on hardware
//
#include <hip/hip_runtime.h>

// ---------------------------------------------------------------------------
// MI455X (gfx1250) implementation.
//  * All GEMMs: v_wmma_f32_16x16x32_bf16 (bf16 in, f32 accumulate).
//  * LSTM input projection hoisted to a parallel GEMM (same xs both layers).
//  * Recurrent step: Whh held in VGPR fragments across the whole time loop.
//  * GEMM bf16 A-tiles staged with GLOBAL_LOAD_ASYNC_TO_LDS_B128 (ASYNCcnt).
//  * Sinkhorn fully LDS-resident per batch matrix (156 KB < 160 KB CU-mode).
// ---------------------------------------------------------------------------

typedef float  v8f    __attribute__((ext_vector_type(8)));
typedef __bf16 v16bf  __attribute__((ext_vector_type(16)));
typedef __bf16 bf16x8 __attribute__((ext_vector_type(8)));
typedef int    v4i    __attribute__((vector_size(16)));   // matches builtin param

#define LEAK 0.01f

#if defined(__AMDGCN__) &&                                            \
    __has_builtin(__builtin_amdgcn_global_load_async_to_lds_b128) && \
    __has_builtin(__builtin_amdgcn_s_wait_asynccnt)
#define USE_ASYNC_LDS 1
typedef __attribute__((address_space(1))) v4i* gptr_v4i;  // global
typedef __attribute__((address_space(3))) v4i* lptr_v4i;  // LDS
#else
#define USE_ASYNC_LDS 0
#endif

__device__ __forceinline__ float sigf(float x) { return 1.0f / (1.0f + __expf(-x)); }

// ---------------------------------------------------------------------------
// Generic tiled GEMM: out[M,Nw] = act(A[M,KT] @ W[Nw,KT]^T + bias1 + bias2)
// Block = 256 threads (8 waves), 64x64 output tile, single K pass (KT<=128).
// Wave w computes row-tile (w>>1), col-tiles {(w&1)*2, (w&1)*2+1}.
// ---------------------------------------------------------------------------
template <int KT, int ACT, bool AF32, bool OBF16>
__global__ __launch_bounds__(256) void gemm_kernel(
    const void* __restrict__ Aptr, const float* __restrict__ W,
    const float* __restrict__ bias1, const float* __restrict__ bias2,
    void* __restrict__ outPtr, int M, int Nw) {
  static_assert(KT == 32 || KT == 128, "KT must be 32 or 128");
  __shared__ __attribute__((aligned(16))) __bf16 Asub[64 * KT];
  __shared__ __attribute__((aligned(16))) __bf16 Wsub[64 * KT];

  const int tid = threadIdx.x;
  const int m0 = blockIdx.x * 64;
  const int n0 = blockIdx.y * 64;

  // ---- stage A tile ----
  if constexpr (AF32) {
    // f32 -> bf16 conversion required: VALU path.
    const float4* A4 =
        reinterpret_cast<const float4*>((const float*)Aptr + (size_t)m0 * KT);
    for (int i = tid; i < 64 * KT / 4; i += 256) {
      float4 v = A4[i];
      __bf16* d = &Asub[i * 4];
      d[0] = (__bf16)v.x; d[1] = (__bf16)v.y;
      d[2] = (__bf16)v.z; d[3] = (__bf16)v.w;
    }
  } else {
    // Raw bf16 copy: async direct-to-LDS (no VGPR round trip, ASYNCcnt).
#if USE_ASYNC_LDS
    const uint4* Ag =
        reinterpret_cast<const uint4*>((const __bf16*)Aptr + (size_t)m0 * KT);
    uint4* Al = reinterpret_cast<uint4*>(Asub);
    for (int i = tid; i < 64 * KT / 8; i += 256) {  // 16B per lane per op
      __builtin_amdgcn_global_load_async_to_lds_b128(
          (gptr_v4i)(Ag + i), (lptr_v4i)(Al + i), 0, 0);
    }
#else
    const uint2* A4 =
        reinterpret_cast<const uint2*>((const __bf16*)Aptr + (size_t)m0 * KT);
    uint2* d = reinterpret_cast<uint2*>(Asub);
    for (int i = tid; i < 64 * KT / 4; i += 256) d[i] = A4[i];
#endif
  }
  // ---- stage W tile (f32 -> bf16, zero-fill rows >= Nw): overlaps with the
  // in-flight async A-tile copies.
  for (int i = tid; i < 64 * KT / 4; i += 256) {
    int row = i / (KT / 4);
    int c4  = i % (KT / 4);
    int gr  = n0 + row;
    float4 v = make_float4(0.f, 0.f, 0.f, 0.f);
    if (gr < Nw)
      v = *reinterpret_cast<const float4*>(W + (size_t)gr * KT + c4 * 4);
    __bf16* d = &Wsub[row * KT + c4 * 4];
    d[0] = (__bf16)v.x; d[1] = (__bf16)v.y;
    d[2] = (__bf16)v.z; d[3] = (__bf16)v.w;
  }
#if USE_ASYNC_LDS
  if constexpr (!AF32) __builtin_amdgcn_s_wait_asynccnt(0);
#endif
  __syncthreads();

  const int lane = tid & 31, hi = lane >> 4, nl = lane & 15;
  const int wv = tid >> 5, tr = wv >> 1, tcb = (wv & 1) * 2;

  v8f acc0 = {}, acc1 = {};
#pragma unroll
  for (int kk = 0; kk < KT / 32; ++kk) {
    // A fragment: K = kk*32 + hi*8 + (j&7) + (j>=8)*16
    const __bf16* ar = &Asub[(tr * 16 + nl) * KT + kk * 32 + hi * 8];
    bf16x8 alo = *reinterpret_cast<const bf16x8*>(ar);
    bf16x8 ahi = *reinterpret_cast<const bf16x8*>(ar + 16);
    v16bf a;
#pragma unroll
    for (int j = 0; j < 8; ++j) { a[j] = alo[j]; a[j + 8] = ahi[j]; }
    // B fragments: K = kk*32 + hi*8 + (j&7) + (j>=8)*16
    const __bf16* b0p = &Wsub[(tcb * 16 + nl) * KT + kk * 32 + hi * 8];
    const __bf16* b1p = &Wsub[((tcb + 1) * 16 + nl) * KT + kk * 32 + hi * 8];
    bf16x8 b0l = *reinterpret_cast<const bf16x8*>(b0p);
    bf16x8 b0h = *reinterpret_cast<const bf16x8*>(b0p + 16);
    bf16x8 b1l = *reinterpret_cast<const bf16x8*>(b1p);
    bf16x8 b1h = *reinterpret_cast<const bf16x8*>(b1p + 16);
    v16bf b0, b1;
#pragma unroll
    for (int j = 0; j < 8; ++j) {
      b0[j] = b0l[j]; b0[j + 8] = b0h[j];
      b1[j] = b1l[j]; b1[j + 8] = b1h[j];
    }
    acc0 = __builtin_amdgcn_wmma_f32_16x16x32_bf16(false, a, false, b0,
                                                   (short)0, acc0, false, false);
    acc1 = __builtin_amdgcn_wmma_f32_16x16x32_bf16(false, a, false, b1,
                                                   (short)0, acc1, false, false);
  }

  // ---- epilogue: bias + activation + store (C layout: M=r+8*hi, N=lane&15)
  const int mbase = m0 + tr * 16 + hi * 8;
#pragma unroll
  for (int tci = 0; tci < 2; ++tci) {
    v8f acc = tci ? acc1 : acc0;
    int n = n0 + (tcb + tci) * 16 + nl;
    if (n < Nw) {
      float bv = bias1[n] + (bias2 ? bias2[n] : 0.0f);
#pragma unroll
      for (int r = 0; r < 8; ++r) {
        float v = acc[r] + bv;
        if (ACT == 1) v = (v >= 0.0f) ? v : LEAK * v;
        size_t idx = (size_t)(mbase + r) * Nw + n;
        if (OBF16) ((__bf16*)outPtr)[idx] = (__bf16)v;
        else       ((float*)outPtr)[idx] = v;
      }
    }
  }
}

// ---------------------------------------------------------------------------
// LSTM layer: 32 blocks x 16 batch rows, 8 waves. h/c in LDS for all 200
// steps; each wave holds its 64-gate-column slice of Whh as 16 bf16 WMMA
// fragments in VGPRs for the whole kernel. XW = precomputed x@Wih^T + bias.
// ---------------------------------------------------------------------------
__global__ __launch_bounds__(256) void lstm_layer_kernel(
    const float* __restrict__ XW,    // [B*200, 512]  (bias folded in)
    const float* __restrict__ Whh,   // [512, 128] f32
    float* __restrict__ h_state, float* __restrict__ c_state,  // [B,128]
    __bf16* __restrict__ hs,         // [B*200, 128] (last layer only)
    int initZero, int writeHs) {
  constexpr int Nseq = 200;
  __shared__ __attribute__((aligned(16))) __bf16 hS[16 * 128];
  __shared__ float cS[16 * 128];
  __shared__ float gS[16 * 512];

  const int tid = threadIdx.x;
  const int b0 = blockIdx.x * 16;
  const int lane = tid & 31, hi = lane >> 4, nl = lane & 15, wv = tid >> 5;

  // init h,c
  for (int i = tid; i < 2048; i += 256) {
    int row = i >> 7, col = i & 127;
    float hv = initZero ? 0.0f : h_state[(size_t)(b0 + row) * 128 + col];
    float cv = initZero ? 0.0f : c_state[(size_t)(b0 + row) * 128 + col];
    hS[i] = (__bf16)hv;
    cS[i] = cv;
  }

  // Preload Whh fragments: wave wv owns gate-col tiles ct = wv*4 .. wv*4+3.
  v16bf Wf[4][4];
#pragma unroll
  for (int i = 0; i < 4; ++i) {
    int col = (wv * 4 + i) * 16 + nl;
    const float* wr = Whh + (size_t)col * 128;
#pragma unroll
    for (int kk = 0; kk < 4; ++kk) {
      const float4* p = reinterpret_cast<const float4*>(wr + kk * 32 + hi * 8);
      float4 x0 = p[0], x1 = p[1], x2 = p[4], x3 = p[5];
      v16bf b;
      b[0]  = (__bf16)x0.x; b[1]  = (__bf16)x0.y; b[2]  = (__bf16)x0.z; b[3]  = (__bf16)x0.w;
      b[4]  = (__bf16)x1.x; b[5]  = (__bf16)x1.y; b[6]  = (__bf16)x1.z; b[7]  = (__bf16)x1.w;
      b[8]  = (__bf16)x2.x; b[9]  = (__bf16)x2.y; b[10] = (__bf16)x2.z; b[11] = (__bf16)x2.w;
      b[12] = (__bf16)x3.x; b[13] = (__bf16)x3.y; b[14] = (__bf16)x3.z; b[15] = (__bf16)x3.w;
      Wf[i][kk] = b;
    }
  }
  __syncthreads();

  for (int t = 0; t < Nseq; ++t) {
    // A fragments from h (16 batch rows x 128)
    v16bf a[4];
#pragma unroll
    for (int kk = 0; kk < 4; ++kk) {
      const __bf16* ar = &hS[nl * 128 + kk * 32 + hi * 8];
      bf16x8 lo = *reinterpret_cast<const bf16x8*>(ar);
      bf16x8 hh = *reinterpret_cast<const bf16x8*>(ar + 16);
#pragma unroll
      for (int j = 0; j < 8; ++j) { a[kk][j] = lo[j]; a[kk][j + 8] = hh[j]; }
    }
#pragma unroll
    for (int i = 0; i < 4; ++i) {
      int ct = wv * 4 + i;
      const float* xwb =
          XW + ((size_t)(b0 + hi * 8) * Nseq + t) * 512 + ct * 16 + nl;
      v8f acc;
#pragma unroll
      for (int r = 0; r < 8; ++r) acc[r] = xwb[(size_t)r * Nseq * 512];
      if (t + 1 < Nseq) __builtin_prefetch(xwb + 512, 0, 0);  // next step's XW
#pragma unroll
      for (int kk = 0; kk < 4; ++kk)
        acc = __builtin_amdgcn_wmma_f32_16x16x32_bf16(
            false, a[kk], false, Wf[i][kk], (short)0, acc, false, false);
#pragma unroll
      for (int r = 0; r < 8; ++r)
        gS[(r + hi * 8) * 512 + ct * 16 + nl] = acc[r];
    }
    __syncthreads();

    // elementwise gate update: i,f,g,o at cols {c, 128+c, 256+c, 384+c}
    for (int i2 = tid; i2 < 2048; i2 += 256) {
      int row = i2 >> 7, col = i2 & 127;
      const float* g = &gS[row * 512];
      float iv = sigf(g[col]);
      float fv = sigf(g[128 + col]);
      float gv = tanhf(g[256 + col]);
      float ov = sigf(g[384 + col]);
      float cv = fv * cS[i2] + iv * gv;
      cS[i2] = cv;
      float hv = ov * tanhf(cv);
      hS[i2] = (__bf16)hv;
      if (writeHs)
        hs[((size_t)(b0 + row) * Nseq + t) * 128 + col] = (__bf16)hv;
    }
    __syncthreads();
  }

  // persist state for next layer
  for (int i = tid; i < 2048; i += 256) {
    int row = i >> 7, col = i & 127;
    h_state[(size_t)(b0 + row) * 128 + col] = (float)hS[i];
    c_state[(size_t)(b0 + row) * 128 + col] = cS[i];
  }
}

// ---------------------------------------------------------------------------
// Sinkhorn: one block per batch matrix; entire 200x200 f32 matrix in LDS.
// In-place on d_out (fc2 wrote M there).
// ---------------------------------------------------------------------------
__global__ __launch_bounds__(256) void sinkhorn_kernel(float* __restrict__ M) {
  __shared__ float Ms[200 * 200];   // 156.25 KB
  const int tid = threadIdx.x;
  const size_t base = (size_t)blockIdx.x * 40000;

  for (int i = tid; i < 40000; i += 256) Ms[i] = __expf(M[base + i]);  // tau=1
  __syncthreads();

  for (int it = 0; it < 5; ++it) {
    if (tid < 200) {  // row normalize
      float s = 0.0f;
      for (int k = 0; k < 200; ++k) s += Ms[tid * 200 + k];
      float inv = 1.0f / s;
      for (int k = 0; k < 200; ++k) Ms[tid * 200 + k] *= inv;
    }
    __syncthreads();
    if (tid < 200) {  // column normalize
      float s = 0.0f;
      for (int r = 0; r < 200; ++r) s += Ms[r * 200 + tid];
      float inv = 1.0f / s;
      for (int r = 0; r < 200; ++r) Ms[r * 200 + tid] *= inv;
    }
    __syncthreads();
  }
  for (int i = tid; i < 40000; i += 256) M[base + i] = Ms[i];
}

// ---------------------------------------------------------------------------
extern "C" void kernel_launch(void* const* d_in, const int* in_sizes, int n_in,
                              void* d_out, int out_size, void* d_ws,
                              size_t ws_size, hipStream_t stream) {
  (void)in_sizes; (void)n_in; (void)out_size; (void)ws_size;
  const int B = 512, Nn = 200, H = 128, E = 128;
  const int Mrows = B * Nn;  // 102400

  const float* x     = (const float*)d_in[0];
  const float* w_emb = (const float*)d_in[1];
  const float* b_emb = (const float*)d_in[2];
  const float* w_ih  = (const float*)d_in[3];  // [2,512,128]
  const float* w_hh  = (const float*)d_in[4];  // [2,512,128]
  const float* b_ih  = (const float*)d_in[5];  // [2,512]
  const float* b_hh  = (const float*)d_in[6];  // [2,512]
  const float* w_fc1 = (const float*)d_in[7];  // [128,128]
  const float* b_fc1 = (const float*)d_in[8];
  const float* w_fc2 = (const float*)d_in[9];  // [200,128]
  const float* b_fc2 = (const float*)d_in[10];
  // d_in[11] = do_round (always 0)

  // workspace layout
  char* ws = (char*)d_ws;
  const size_t OFF_XW = 0;                                 // 102400*512*4
  const size_t OFF_E  = OFF_XW + (size_t)Mrows * 512 * 4;  // bf16 102400*128
  const size_t OFF_HS = OFF_E + (size_t)Mrows * E * 2;
  const size_t OFF_O1 = OFF_HS + (size_t)Mrows * H * 2;
  const size_t OFF_H  = OFF_O1 + (size_t)Mrows * E * 2;
  const size_t OFF_C  = OFF_H + (size_t)B * H * 4;
  float*  XW  = (float*)(ws + OFF_XW);
  __bf16* e   = (__bf16*)(ws + OFF_E);
  __bf16* hs  = (__bf16*)(ws + OFF_HS);
  __bf16* o1  = (__bf16*)(ws + OFF_O1);
  float*  hst = (float*)(ws + OFF_H);
  float*  cst = (float*)(ws + OFF_C);
  float*  Mout = (float*)d_out;

  const int mb = Mrows / 64;  // 1600 row blocks

  // 1) embed: e = leaky_relu(x @ w_emb^T + b_emb)   [102400,32]x[128,32]^T
  gemm_kernel<32, 1, true, true><<<dim3(mb, 2), 256, 0, stream>>>(
      x, w_emb, b_emb, nullptr, e, Mrows, 128);

  // 2) per layer: XW = e @ w_ih[l]^T + (b_ih+b_hh);  then recurrent scan
  for (int l = 0; l < 2; ++l) {
    gemm_kernel<128, 0, false, false><<<dim3(mb, 8), 256, 0, stream>>>(
        e, w_ih + (size_t)l * 512 * 128, b_ih + (size_t)l * 512,
        b_hh + (size_t)l * 512, XW, Mrows, 512);
    lstm_layer_kernel<<<B / 16, 256, 0, stream>>>(
        XW, w_hh + (size_t)l * 512 * 128, hst, cst, hs,
        /*initZero=*/(l == 0) ? 1 : 0, /*writeHs=*/(l == 1) ? 1 : 0);
  }

  // 3) fc1: o1 = leaky_relu(hs @ w_fc1^T + b_fc1)
  gemm_kernel<128, 1, false, true><<<dim3(mb, 2), 256, 0, stream>>>(
      hs, w_fc1, b_fc1, nullptr, o1, Mrows, 128);

  // 4) fc2: M = o1 @ w_fc2^T + b_fc2  (Nw=200, padded col blocks masked)
  gemm_kernel<128, 0, false, false><<<dim3(mb, 4), 256, 0, stream>>>(
      o1, w_fc2, b_fc2, nullptr, Mout, Mrows, 200);

  // 5) Sinkhorn, in place on d_out
  sinkhorn_kernel<<<B, 256, 0, stream>>>(Mout);
}
